// NeoX20BModel_13494787244058
// MI455X (gfx1250) — matrix operand order repaired and checked by reading, not compiled, not run on hardware
//
#include <hip/hip_runtime.h>
#include <hip/hip_bf16.h>
#include <math.h>

// ---------------------------------------------------------------------------
// GPT-NeoX-20B-style forward (L=2, B=1, S=2048, H=2048, NH=16, HD=128,
// ROT=32, FF=8192, V=32000) for MI455X (gfx1250, wave32, WMMA + TDM).
// GEMMs: f32 -> bf16 on the fly, v_wmma_f32_16x16x32_bf16, f32 accumulate.
// A tiles DMA'd to LDS by the Tensor Data Mover, double-buffered so the DMA
// of tile i+1 overlaps the 64 WMMAs of tile i (wait tensorcnt<=1).
// B tiles staged transposed so B-frag reads are contiguous ds_load_b128.
// ---------------------------------------------------------------------------

typedef __attribute__((ext_vector_type(16))) __bf16 v16bf;
typedef __attribute__((ext_vector_type(8)))  float  v8f;
typedef __attribute__((ext_vector_type(4)))  unsigned int u32x4;
typedef __attribute__((ext_vector_type(8)))  int    i32x8;
typedef __attribute__((ext_vector_type(4)))  int    i32x4;

#define S_LEN  2048
#define HID    2048
#define NHEAD  16
#define HDIM   128
#define QKVW   (3 * HID)      // 6144
#define HEADW  (3 * HDIM)     // 384 per head inside qkv row
#define FFD    8192
#define NLAYER 2
#define VOCAB  32000

#define BM 128
#define BN 128
#define BK 32
#define APAD 4                // A tile LDS row = 36 dwords (TDM pad feature)
#define BPAD 16               // B tile LDS row = 48 bf16 = 96B (16B multiple)

// ---------------------------------------------------------------------------
// Embedding gather: h[s,:] = embed[x[s],:]
// ---------------------------------------------------------------------------
__global__ __launch_bounds__(256) void embed_kernel(const int* __restrict__ x,
                                                    const float* __restrict__ embed,
                                                    float* __restrict__ h) {
  int s = blockIdx.x;
  int t = threadIdx.x;
  int tok = x[s];
  const float* src = embed + (size_t)tok * HID;
  float* dst = h + (size_t)s * HID;
#pragma unroll
  for (int i = 0; i < HID / 256; ++i) dst[t + i * 256] = src[t + i * 256];
}

// ---------------------------------------------------------------------------
// LayerNorm, one 256-thread block per row of H=2048.
// ---------------------------------------------------------------------------
__global__ __launch_bounds__(256) void layernorm_kernel(const float* __restrict__ x,
                                                        const float* __restrict__ g,
                                                        const float* __restrict__ b,
                                                        float* __restrict__ y) {
  __shared__ float rs[256];
  __shared__ float rs2[256];
  int row = blockIdx.x;
  int t = threadIdx.x;
  const float* xr = x + (size_t)row * HID;
  float v[HID / 256];
  float s = 0.f, s2 = 0.f;
#pragma unroll
  for (int i = 0; i < HID / 256; ++i) {
    v[i] = xr[t + i * 256];
    s += v[i];
    s2 += v[i] * v[i];
  }
  rs[t] = s; rs2[t] = s2;
  __syncthreads();
  for (int off = 128; off > 0; off >>= 1) {
    if (t < off) { rs[t] += rs[t + off]; rs2[t] += rs2[t + off]; }
    __syncthreads();
  }
  float mean = rs[0] * (1.f / HID);
  float var  = rs2[0] * (1.f / HID) - mean * mean;
  float inv  = rsqrtf(var + 1e-5f);
  float* yr = y + (size_t)row * HID;
#pragma unroll
  for (int i = 0; i < HID / 256; ++i) {
    int c = t + i * 256;
    yr[c] = (v[i] - mean) * inv * g[c] + b[c];
  }
}

// ---------------------------------------------------------------------------
// Rotary embedding, in place on qkv [S, 6144]. 256 threads = 16 heads x 16 dims.
// ---------------------------------------------------------------------------
__global__ __launch_bounds__(256) void rotary_kernel(float* __restrict__ qkv) {
  int s = blockIdx.x;
  int t = threadIdx.x;
  int head = t >> 4;
  int d = t & 15;
  float ang = (float)s * __expf(-((float)(2 * d) / 32.f) * 9.210340371976184f);
  float c = __cosf(ang), sn = __sinf(ang);
  float* base = qkv + (size_t)s * QKVW + head * HEADW;
  float a = base[d], b2 = base[d + 16];
  base[d]      = a * c - b2 * sn;
  base[d + 16] = b2 * c + a * sn;
  float* kb = base + HDIM;
  a = kb[d]; b2 = kb[d + 16];
  kb[d]      = a * c - b2 * sn;
  kb[d + 16] = b2 * c + a * sn;
}

// ---------------------------------------------------------------------------
// Tiled bf16 WMMA GEMM: C[M,N] = A[M,K] * B[K,N] (+bias) (+gelu)
// BM=128 x BN=128 per 256-thread block; 8 waves in 4(M) x 2(N) grid; each wave
// owns a 32x64 tile = 2x4 16x16 frags -> 8 WMMAs per K-step per wave.
// A tile: double-buffered TDM DMA (overlapped with compute via tensorcnt<=1).
// B tile: double-buffered cooperative transposed bf16 staging.
// ---------------------------------------------------------------------------
__global__ __launch_bounds__(256) void gemm_kernel(const float* __restrict__ A,
                                                   const float* __restrict__ B,
                                                   const float* __restrict__ bias,
                                                   float* __restrict__ C,
                                                   int M, int N, int K, int act_gelu) {
  __shared__ __align__(16) float  AsF[2][BM][BK + APAD];  // 2 x 128 x 36 f32 (TDM)
  __shared__ __align__(16) __bf16 BsT[2][BN][BK + BPAD];  // 2 x 128 x 48 bf16

  int t = threadIdx.x;
  int bm = blockIdx.y * BM;
  int bn = blockIdx.x * BN;
  int wave = t >> 5;
  int lane = t & 31;
  int hf = lane >> 4;       // wave half (0: lanes 0-15, 1: lanes 16-31)
  int l15 = lane & 15;
  int kh = hf * 16;         // A-frag K sub-offset
  int wm = (wave & 3) * 32; // wave tile origin in block tile
  int wn = (wave >> 2) * 64;

  // --- TDM descriptor (D#) for the A tile: 32(x) x 128(y) f32, stride K ----
  unsigned lds_a0 = (unsigned)(uintptr_t)&AsF[0][0][0];
  unsigned lds_a1 = (unsigned)(uintptr_t)&AsF[1][0][0];
  unsigned long long gbase =
      (unsigned long long)(uintptr_t)A + (unsigned long long)bm * (unsigned)K * 4ull;
  u32x4 g0;
  g0[0] = 1u;          // count=1, user descriptor
  g0[1] = 0u;          // LDS byte address (per-issue)
  g0[2] = 0u;          // global addr lo (per-issue)
  g0[3] = 0u;          // global addr hi | type=2<<30 (per-issue)
  i32x8 g1;            // data_size=4B, pad 32 dwords -> +4 dwords, dims/strides
  g1[0] = (int)((2u << 16) | (1u << 20) | (4u << 22) | (3u << 25));
  g1[1] = (int)(((unsigned)K & 0xFFFFu) << 16);                    // td0 lo
  g1[2] = (int)(((unsigned)K >> 16) | (((unsigned)M & 0xFFFFu) << 16)); // td0 hi|td1 lo
  g1[3] = (int)(((unsigned)M >> 16) | (32u << 16));                // td1 hi|tile_dim0=32
  g1[4] = 128;                                                     // tile_dim1=128
  g1[5] = (int)(unsigned)K;                                        // stride0 lo
  g1[6] = 0;
  g1[7] = 0;
  i32x4 gz4;
  gz4[0] = 0; gz4[1] = 0; gz4[2] = 0; gz4[3] = 0;
  i32x8 gz8;
#pragma unroll
  for (int i = 0; i < 8; ++i) gz8[i] = 0;

  v8f acc[2][4];
#pragma unroll
  for (int a = 0; a < 2; ++a)
#pragma unroll
    for (int b = 0; b < 4; ++b)
#pragma unroll
      for (int i = 0; i < 8; ++i) acc[a][b][i] = 0.f;

  int col = t & 127;         // B staging: this thread's column
  int kq = (t >> 7) * 16;    // and K sub-range [kq, kq+16)
  // hoisted base pointer for B staging; advances by BK rows per tile
  const float* bstage = B + (size_t)kq * N + bn + col;
  const size_t bstep = (size_t)BK * N;

  int ntiles = K / BK;

  // ---- prologue: stage tile 0 (B cooperative, A via TDM) -------------------
  {
    const float* bp = bstage;
    union { __bf16 h[8]; uint4 v; } pk;
#pragma unroll
    for (int half2 = 0; half2 < 2; ++half2) {
#pragma unroll
      for (int kk = 0; kk < 8; ++kk) {
        pk.h[kk] = (__bf16)(*bp);
        bp += N;
      }
      *(uint4*)&BsT[0][col][kq + half2 * 8] = pk.v;
    }
  }
  if (t < 32) {
    g0[1] = lds_a0;
    g0[2] = (unsigned)gbase;
    g0[3] = (unsigned)(gbase >> 32) | 0x80000000u;  // addr[56:32] | type=2
    __builtin_amdgcn_tensor_load_to_lds(g0, g1, gz4, gz4, gz8, 0);
  }

  for (int i = 0; i < ntiles; ++i) {
    int cur = i & 1;
    int nxt = cur ^ 1;
    // ---- stage tile i+1 into the other buffer (overlaps with TDM(i)) ------
    if (i + 1 < ntiles) {
      const float* bp = bstage + (size_t)(i + 1) * bstep;
      __builtin_prefetch(bp + bstep, 0, 1);
      union { __bf16 h[8]; uint4 v; } pk;
#pragma unroll
      for (int half2 = 0; half2 < 2; ++half2) {
#pragma unroll
        for (int kk = 0; kk < 8; ++kk) {
          pk.h[kk] = (__bf16)(*bp);
          bp += N;
        }
        *(uint4*)&BsT[nxt][col][kq + half2 * 8] = pk.v;
      }
    }
    if (t < 32) {
      if (i + 1 < ntiles) {
        unsigned long long ga = gbase + (unsigned long long)((i + 1) * BK) * 4ull;
        g0[1] = (nxt ? lds_a1 : lds_a0);
        g0[2] = (unsigned)ga;
        g0[3] = (unsigned)(ga >> 32) | 0x80000000u;
        __builtin_amdgcn_tensor_load_to_lds(g0, g1, gz4, gz4, gz8, 0);
        // TDM completes in order: <=1 outstanding means TDM(i) has landed,
        // TDM(i+1) still streaming while we compute tile i.
        __builtin_amdgcn_s_wait_tensorcnt(1);
      } else {
        __builtin_amdgcn_s_wait_tensorcnt(0);
      }
    }
    __syncthreads();

    // A frags: lane half 0 holds K 0-15, half 1 holds K 16-31
    v16bf af[2];
#pragma unroll
    for (int fm = 0; fm < 2; ++fm) {
      const float* p = &AsF[cur][wm + fm * 16 + l15][0];
#pragma unroll
      for (int e = 0; e < 8; ++e) {
        af[fm][e] = (__bf16)p[kh + e];
        af[fm][8 + e] = (__bf16)p[kh + 8 + e];
      }
    }
    // B frags: col N = lane&15; half 0 holds K 0-15, half 1 holds K 16-31.
    v16bf bfr[4];
#pragma unroll
    for (int fn = 0; fn < 4; ++fn) {
      const __bf16* p = &BsT[cur][wn + fn * 16 + l15][hf * 16];
#pragma unroll
      for (int e = 0; e < 16; ++e) bfr[fn][e] = p[e];
    }
#pragma unroll
    for (int fm = 0; fm < 2; ++fm)
#pragma unroll
      for (int fn = 0; fn < 4; ++fn)
        acc[fm][fn] = __builtin_amdgcn_wmma_f32_16x16x32_bf16(
            false, af[fm], false, bfr[fn], (short)0, acc[fm][fn], false, false);
    __syncthreads();
  }

  // Epilogue: C/D frag: VGPR i -> row i + 8*half; col = lane&15
#pragma unroll
  for (int fm = 0; fm < 2; ++fm)
#pragma unroll
    for (int fn = 0; fn < 4; ++fn) {
      int c = bn + wn + fn * 16 + l15;
      float bv = bias ? bias[c] : 0.f;
#pragma unroll
      for (int i = 0; i < 8; ++i) {
        int row = bm + wm + fm * 16 + hf * 8 + i;
        float x = acc[fm][fn][i] + bv;
        if (act_gelu) {
          float x3 = x * (1.f + 0.044715f * x * x);
          x = x * 0.5f * (1.f + tanhf(0.79788456f * x3));
        }
        C[(size_t)row * N + c] = x;
      }
    }
}

// ---------------------------------------------------------------------------
// Flash attention: one wave per (16-query block, head). Causal, online softmax.
// scores: Q(16x128 bf16, pre-scaled) x K^T via 8 WMMAs per 32 keys,
// P routed through LDS (C-layout -> A-layout), ctx += P*V via 8 WMMAs.
// ---------------------------------------------------------------------------
__global__ __launch_bounds__(32) void flash_attn_kernel(const float* __restrict__ qkv,
                                                        float* __restrict__ ctx_out) {
  __shared__ __bf16 P[16][32];
  int qb = blockIdx.x;
  int head = blockIdx.y;
  int lane = threadIdx.x;
  int hf = lane >> 4;
  int l15 = lane & 15;
  int kh = hf * 16;
  const float scale = 0.08838834764831845f; // 1/sqrt(128)

  v16bf qf[4];
  {
    const float* qrow = qkv + (size_t)(qb * 16 + l15) * QKVW + head * HEADW;
#pragma unroll
    for (int kk = 0; kk < 4; ++kk) {
#pragma unroll
      for (int e = 0; e < 8; ++e) {
        qf[kk][e] = (__bf16)(qrow[kk * 32 + kh + e] * scale);
        qf[kk][8 + e] = (__bf16)(qrow[kk * 32 + kh + 8 + e] * scale);
      }
    }
  }

  v8f ctx[8];
#pragma unroll
  for (int nn = 0; nn < 8; ++nn)
#pragma unroll
    for (int i = 0; i < 8; ++i) ctx[nn][i] = 0.f;
  float m[8], l[8];
#pragma unroll
  for (int i = 0; i < 8; ++i) { m[i] = -3.0e38f; l[i] = 0.f; }

  int qrow0 = qb * 16 + hf * 8;
  int kbmax = (qb * 16 + 15) >> 5;
  for (int kb = 0; kb <= kbmax; ++kb) {
    int k0 = kb * 32;
    v8f s0, s1;
#pragma unroll
    for (int i = 0; i < 8; ++i) { s0[i] = 0.f; s1[i] = 0.f; }

    const float* kbase0 = qkv + (size_t)(k0 + l15) * QKVW + head * HEADW + HDIM;
    const float* kbase1 = kbase0 + (size_t)16 * QKVW;
#pragma unroll
    for (int kk = 0; kk < 4; ++kk) {
      const float* p0 = kbase0 + kk * 32 + hf * 16;
      const float* p1 = kbase1 + kk * 32 + hf * 16;
      v16bf b0, b1;
#pragma unroll
      for (int e = 0; e < 16; ++e) {
        b0[e] = (__bf16)p0[e];
        b1[e] = (__bf16)p1[e];
      }
      s0 = __builtin_amdgcn_wmma_f32_16x16x32_bf16(false, qf[kk], false, b0,
                                                   (short)0, s0, false, false);
      s1 = __builtin_amdgcn_wmma_f32_16x16x32_bf16(false, qf[kk], false, b1,
                                                   (short)0, s1, false, false);
    }

#pragma unroll
    for (int i = 0; i < 8; ++i) {
      int qi = qrow0 + i;
      float v0 = (k0 + l15 > qi) ? -10000.f : s0[i];
      float v1 = (k0 + 16 + l15 > qi) ? -10000.f : s1[i];
      float rmax = fmaxf(v0, v1);
#pragma unroll
      for (int off = 8; off > 0; off >>= 1)
        rmax = fmaxf(rmax, __shfl_xor(rmax, off, 16));
      float mn = fmaxf(m[i], rmax);
      float corr = __expf(m[i] - mn);
      float p0 = __expf(v0 - mn);
      float p1 = __expf(v1 - mn);
      float rsum = p0 + p1;
#pragma unroll
      for (int off = 8; off > 0; off >>= 1) rsum += __shfl_xor(rsum, off, 16);
      l[i] = l[i] * corr + rsum;
      m[i] = mn;
#pragma unroll
      for (int nn = 0; nn < 8; ++nn) ctx[nn][i] *= corr;
      P[hf * 8 + i][l15] = (__bf16)p0;
      P[hf * 8 + i][16 + l15] = (__bf16)p1;
    }
    __syncthreads();

    v16bf pf;
    {
      const __bf16* pr = &P[l15][0];
#pragma unroll
      for (int e = 0; e < 8; ++e) {
        pf[e] = pr[kh + e];
        pf[8 + e] = pr[kh + 8 + e];
      }
    }
    // V tile: per-lane pointer walks down keys with stride QKVW (hoisted).
    const float* vcol0 = qkv + (size_t)(k0 + hf * 16) * QKVW + head * HEADW
                         + 2 * HDIM + l15;
#pragma unroll
    for (int nn = 0; nn < 8; ++nn) {
      const float* vp = vcol0 + nn * 16;
      v16bf vb;
#pragma unroll
      for (int e = 0; e < 16; ++e) {
        vb[e] = (__bf16)(*vp);
        vp += QKVW;
      }
      ctx[nn] = __builtin_amdgcn_wmma_f32_16x16x32_bf16(false, pf, false, vb,
                                                        (short)0, ctx[nn], false, false);
    }
    __syncthreads();
  }

#pragma unroll
  for (int nn = 0; nn < 8; ++nn) {
    int c = head * HDIM + nn * 16 + l15;
#pragma unroll
    for (int i = 0; i < 8; ++i) {
      int row = qb * 16 + hf * 8 + i;
      ctx_out[(size_t)row * HID + c] = ctx[nn][i] / l[i];
    }
  }
}

// ---------------------------------------------------------------------------
// h = h + attn_out + mlp_out
// ---------------------------------------------------------------------------
__global__ __launch_bounds__(256) void residual_kernel(float* __restrict__ h,
                                                       const float* __restrict__ a,
                                                       const float* __restrict__ b,
                                                       int n) {
  int i = blockIdx.x * 256 + threadIdx.x;
  if (i < n) h[i] = h[i] + a[i] + b[i];
}

// ---------------------------------------------------------------------------
extern "C" void kernel_launch(void* const* d_in, const int* in_sizes, int n_in,
                              void* d_out, int out_size, void* d_ws, size_t ws_size,
                              hipStream_t stream) {
  const int*   x       = (const int*)d_in[0];
  const float* embed   = (const float*)d_in[1];
  const float* ln1_g   = (const float*)d_in[2];
  const float* ln1_b   = (const float*)d_in[3];
  const float* ln2_g   = (const float*)d_in[4];
  const float* ln2_b   = (const float*)d_in[5];
  const float* qkv_w   = (const float*)d_in[6];
  const float* qkv_b   = (const float*)d_in[7];
  const float* dense_w = (const float*)d_in[8];
  const float* dense_b = (const float*)d_in[9];
  const float* fc1_w   = (const float*)d_in[10];
  const float* fc1_b   = (const float*)d_in[11];
  const float* fc2_w   = (const float*)d_in[12];
  const float* fc2_b   = (const float*)d_in[13];
  const float* fln_g   = (const float*)d_in[14];
  const float* fln_b   = (const float*)d_in[15];
  const float* logitsw = (const float*)d_in[16];
  float* out = (float*)d_out;

  float* ws   = (float*)d_ws;
  float* h    = ws;                               // S*H
  float* ln   = h + (size_t)S_LEN * HID;          // S*H
  float* qkv  = ln + (size_t)S_LEN * HID;         // S*3H
  float* ctx  = qkv + (size_t)S_LEN * QKVW;       // S*H
  float* attn = ctx + (size_t)S_LEN * HID;        // S*H
  float* ff   = attn + (size_t)S_LEN * HID;       // S*FF
  float* mlp  = ff + (size_t)S_LEN * FFD;         // S*H

  embed_kernel<<<S_LEN, 256, 0, stream>>>(x, embed, h);

  for (int layer = 0; layer < NLAYER; ++layer) {
    // ---- attention branch ----
    layernorm_kernel<<<S_LEN, 256, 0, stream>>>(h, ln1_g + layer * HID,
                                                ln1_b + layer * HID, ln);
    gemm_kernel<<<dim3(QKVW / BN, S_LEN / BM), 256, 0, stream>>>(
        ln, qkv_w + (size_t)layer * HID * QKVW, qkv_b + (size_t)layer * QKVW,
        qkv, S_LEN, QKVW, HID, 0);
    rotary_kernel<<<S_LEN, 256, 0, stream>>>(qkv);
    flash_attn_kernel<<<dim3(S_LEN / 16, NHEAD), 32, 0, stream>>>(qkv, ctx);
    gemm_kernel<<<dim3(HID / BN, S_LEN / BM), 256, 0, stream>>>(
        ctx, dense_w + (size_t)layer * HID * HID, dense_b + (size_t)layer * HID,
        attn, S_LEN, HID, HID, 0);

    // ---- mlp branch (reads pre-attention h, like the reference) ----
    layernorm_kernel<<<S_LEN, 256, 0, stream>>>(h, ln2_g + layer * HID,
                                                ln2_b + layer * HID, ln);
    gemm_kernel<<<dim3(FFD / BN, S_LEN / BM), 256, 0, stream>>>(
        ln, fc1_w + (size_t)layer * HID * FFD, fc1_b + (size_t)layer * FFD,
        ff, S_LEN, FFD, HID, 1 /*gelu*/);
    gemm_kernel<<<dim3(HID / BN, S_LEN / BM), 256, 0, stream>>>(
        ff, fc2_w + (size_t)layer * FFD * HID, fc2_b + (size_t)layer * HID,
        mlp, S_LEN, HID, FFD, 0);

    residual_kernel<<<(S_LEN * HID + 255) / 256, 256, 0, stream>>>(
        h, attn, mlp, S_LEN * HID);
  }

  layernorm_kernel<<<S_LEN, 256, 0, stream>>>(h, fln_g, fln_b, ln);
  gemm_kernel<<<dim3(VOCAB / BN, S_LEN / BM), 256, 0, stream>>>(
      ln, logitsw, nullptr, out, S_LEN, VOCAB, HID, 0);
}
